// CircleLoss_21277267984384
// MI455X (gfx1250) — compile-verified, matrix operand order checked by reading, not run on hardware
//
#include <hip/hip_runtime.h>

// ---------------------------------------------------------------------------
// CircleLoss on MI455X (gfx1250): fused normalize -> f@f^T (WMMA f16) ->
// masked online logsumexp (base-2, native v_exp_f32) -> softplus mean.
// sim matrix is never materialized (saves 2x256MB of HBM traffic).
// ---------------------------------------------------------------------------

typedef __attribute__((ext_vector_type(4)))  _Float16 v4h;
typedef __attribute__((ext_vector_type(8)))  _Float16 v8h;
typedef __attribute__((ext_vector_type(16))) _Float16 v16h;
typedef __attribute__((ext_vector_type(8)))  float    v8f;

#define BATCH      8192
#define DIM        128
#define COL_SPLIT  4          // column chunks per row strip (parallelism)
#define COLS_PER_CHUNK (BATCH / COL_SPLIT)      // 2048
#define TILES_PER_CHUNK (COLS_PER_CHUNK / 16)   // 128

// Base-2 scaled constants: logit2 = logit * log2(e), so v_exp_f32 applies raw.
#define GAMMA_LOG2E 369.32993046f   // 256 * log2(e)
#define LN2         0.69314718056f
#define O_P      1.25f
#define DELTA_P  0.75f
#define DELTA_N  0.25f
#define NEG_BIG  -1e30f

// --------------------------------------------------------------------------
// Kernel 1: L2-normalize rows of feat (f32) into an f16 copy for WMMA.
// One wave per row: 32 lanes x float4.
// --------------------------------------------------------------------------
__global__ void circle_normalize(const float* __restrict__ feat,
                                 _Float16* __restrict__ f16) {
  const int lane = threadIdx.x & 31;
  const int wave = threadIdx.x >> 5;
  const int row  = blockIdx.x * 8 + wave;

  const float4 v = *(const float4*)(feat + (size_t)row * DIM + lane * 4);
  float ss = v.x * v.x + v.y * v.y + v.z * v.z + v.w * v.w;
#pragma unroll
  for (int m = 1; m < 32; m <<= 1) ss += __shfl_xor(ss, m, 32);

  const float scale = 1.0f / fmaxf(sqrtf(ss), 1e-12f);
  v4h o;
  o[0] = (_Float16)(v.x * scale);
  o[1] = (_Float16)(v.y * scale);
  o[2] = (_Float16)(v.z * scale);
  o[3] = (_Float16)(v.w * scale);
  *(v4h*)(f16 + (size_t)row * DIM + lane * 4) = o;
}

// --------------------------------------------------------------------------
// Fragment loader for V_WMMA_F32_16X16X32_F16 (A and B layouts are mirror
// images; for f@f^T both load the same way).  Per §7.12.2:
//   lanes 0-15  : index = lane,   K = kk*32 + {0..7, 16..23}
//   lanes 16-31 : index = lane-16, K = kk*32 + {8..15, 24..31}
// Two 16-byte aligned B128 loads per fragment.
// --------------------------------------------------------------------------
__device__ __forceinline__ v16h load_frag(const _Float16* __restrict__ p,
                                          int row, int kk, int hiHalf) {
  const _Float16* q = p + (size_t)row * DIM + kk * 32 + hiHalf * 8;
  const v8h lo = *(const v8h*)(q);
  const v8h hi = *(const v8h*)(q + 16);
  v16h out;
#pragma unroll
  for (int i = 0; i < 8; ++i) { out[i] = lo[i]; out[i + 8] = hi[i]; }
  return out;
}

// Base-2 online LSE update.  Common case (x <= m) costs 1 v_exp_f32 + fmac;
// the rescale branch decays like 1/t per lane.  Safe for m = NEG_BIG
// (exp2(-1e30) underflows to 0, giving s=1 on first insert).
__device__ __forceinline__ void lse2_insert(float& m, float& s, float x) {
  if (x > m) {
    s = s * __builtin_amdgcn_exp2f(m - x) + 1.0f;
    m = x;
  } else {
    s += __builtin_amdgcn_exp2f(x - m);
  }
}

// Base-2 LSE state merge: (m,s) <- combine((m,s),(om,os)).
__device__ __forceinline__ void lse2_merge(float& m, float& s,
                                           float om, float os) {
  const float nm = fmaxf(m, om);
  s = s * __builtin_amdgcn_exp2f(m - nm) + os * __builtin_amdgcn_exp2f(om - nm);
  m = nm;
}

// --------------------------------------------------------------------------
// Kernel 2: fused GEMM + masked online logsumexp (base-2).
// Wave = one 16-row strip x one 2048-column chunk.
// Block = 4 waves sharing the same A rows (L0 reuse), 512 blocks total.
// Writes per-(row, chunk) partial states (maxP2,sumP,maxN2,sumN).
// --------------------------------------------------------------------------
__global__ void circle_main(const _Float16* __restrict__ f16,
                            const int* __restrict__ label,
                            float* __restrict__ part) {
  const int lane     = threadIdx.x & 31;
  const int wave     = threadIdx.x >> 5;
  const int strip    = blockIdx.x * 4 + wave;    // 0..2047
  const int rowTile  = strip >> 2;               // 0..511
  const int colChunk = strip & 3;                // 0..3
  const int rowBase  = rowTile * 16;
  const int hiHalf   = lane >> 4;                // 0 or 1
  const int lane15   = lane & 15;
  const int hi8      = hiHalf * 8;

  // Resident A fragments: this wave's 16 rows, full K=128 (4 k-steps).
  const int aRow = rowBase + lane15;
  v16h a0 = load_frag(f16, aRow, 0, hiHalf);
  v16h a1 = load_frag(f16, aRow, 1, hiHalf);
  v16h a2 = load_frag(f16, aRow, 2, hiHalf);
  v16h a3 = load_frag(f16, aRow, 3, hiHalf);

  // Labels for the 8 C-layout row slots this lane owns.
  int rLab[8];
#pragma unroll
  for (int r = 0; r < 8; ++r) rLab[r] = label[rowBase + hi8 + r];

  float mP[8], sP[8], mN[8], sN[8];
#pragma unroll
  for (int r = 0; r < 8; ++r) {
    mP[r] = NEG_BIG; sP[r] = 0.0f;
    mN[r] = NEG_BIG; sN[r] = 0.0f;
  }

  const int colStart = colChunk * COLS_PER_CHUNK;
  for (int ct = 0; ct < TILES_PER_CHUNK; ++ct) {
    const int colBase = colStart + ct * 16;
    const int gcol    = colBase + lane15;
    const int cLab    = label[gcol];

    // Prefetch next tile's B rows into cache while this tile computes.
    if (ct + 1 < TILES_PER_CHUNK)
      __builtin_prefetch(f16 + (size_t)(gcol + 16) * DIM, 0, 1);

    v8f acc = {};
    {
      v16h b = load_frag(f16, gcol, 0, hiHalf);
      acc = __builtin_amdgcn_wmma_f32_16x16x32_f16(false, a0, false, b,
                                                   (short)0, acc, false, false);
    }
    {
      v16h b = load_frag(f16, gcol, 1, hiHalf);
      acc = __builtin_amdgcn_wmma_f32_16x16x32_f16(false, a1, false, b,
                                                   (short)0, acc, false, false);
    }
    {
      v16h b = load_frag(f16, gcol, 2, hiHalf);
      acc = __builtin_amdgcn_wmma_f32_16x16x32_f16(false, a2, false, b,
                                                   (short)0, acc, false, false);
    }
    {
      v16h b = load_frag(f16, gcol, 3, hiHalf);
      acc = __builtin_amdgcn_wmma_f32_16x16x32_f16(false, a3, false, b,
                                                   (short)0, acc, false, false);
    }

    // Epilogue: C layout -> slot r holds (row = rowBase+hi8+r, col = gcol).
#pragma unroll
    for (int r = 0; r < 8; ++r) {
      const float sim  = acc[r];
      const int   gr   = rowBase + hi8 + r;
      const bool  same = (rLab[r] == cLab);

      if (same) {
        if (gr != gcol) {  // positive pair (exclude diagonal); rare branch
          const float ap  = fmaxf(O_P - sim, 0.0f);
          const float lp2 = -GAMMA_LOG2E * ap * (sim - DELTA_P);
          lse2_insert(mP[r], sP[r], lp2);
        }
      } else {             // negative pair; dominant path
        const float an  = fmaxf(sim + 0.25f, 0.0f);   // relu(sim - O_n)
        const float ln2 = GAMMA_LOG2E * an * (sim - DELTA_N);
        lse2_insert(mN[r], sN[r], ln2);
      }
    }
  }

  // Butterfly LSE-merge across the 16 columns held by each lane half.
#pragma unroll
  for (int r = 0; r < 8; ++r) {
#pragma unroll
    for (int m = 1; m < 16; m <<= 1) {
      float om = __shfl_xor(mP[r], m, 32);
      float os = __shfl_xor(sP[r], m, 32);
      lse2_merge(mP[r], sP[r], om, os);
      om = __shfl_xor(mN[r], m, 32);
      os = __shfl_xor(sN[r], m, 32);
      lse2_merge(mN[r], sN[r], om, os);
    }
  }

  // Lanes 0 and 16 write the 16 rows' partial states for this chunk.
  if (lane15 == 0) {
#pragma unroll
    for (int r = 0; r < 8; ++r) {
      const int gr = rowBase + hi8 + r;
      float4 pv = {mP[r], sP[r], mN[r], sN[r]};
      *(float4*)(part + ((size_t)gr * COL_SPLIT + colChunk) * 4) = pv;
    }
  }
}

// --------------------------------------------------------------------------
// Kernel 3: merge column-chunk partials per row, softplus, masked mean.
// Single workgroup; no atomics, no init pass needed.
// --------------------------------------------------------------------------
__global__ void circle_finalize(const float* __restrict__ part,
                                float* __restrict__ out) {
  __shared__ float sSum[256];
  __shared__ int   sCnt[256];

  float accS = 0.0f;
  int   accC = 0;
  for (int row = threadIdx.x; row < BATCH; row += 256) {
    float mPt = NEG_BIG, sPt = 0.0f, mNt = NEG_BIG, sNt = 0.0f;
#pragma unroll
    for (int c = 0; c < COL_SPLIT; ++c) {
      const float4 v = *(const float4*)(part + ((size_t)row * COL_SPLIT + c) * 4);
      lse2_merge(mPt, sPt, v.x, v.y);
      lse2_merge(mNt, sNt, v.z, v.w);
    }
    if (sPt > 0.0f && sNt > 0.0f) {  // row has >=1 positive and >=1 negative
      // lse (natural log) = ln2 * (max2 + log2(sum));  z = lse_p + lse_n
      const float z = LN2 * (mPt + __builtin_amdgcn_logf(sPt) +
                             mNt + __builtin_amdgcn_logf(sNt));
      accS += fmaxf(z, 0.0f) + log1pf(__expf(-fabsf(z)));  // stable softplus
      accC += 1;
    }
  }
  sSum[threadIdx.x] = accS;
  sCnt[threadIdx.x] = accC;
  __syncthreads();
  for (int off = 128; off > 0; off >>= 1) {
    if ((int)threadIdx.x < off) {
      sSum[threadIdx.x] += sSum[threadIdx.x + off];
      sCnt[threadIdx.x] += sCnt[threadIdx.x + off];
    }
    __syncthreads();
  }
  if (threadIdx.x == 0) {
    const int n = sCnt[0] > 1 ? sCnt[0] : 1;
    out[0] = sSum[0] / (float)n;
  }
}

// --------------------------------------------------------------------------
extern "C" void kernel_launch(void* const* d_in, const int* in_sizes, int n_in,
                              void* d_out, int out_size, void* d_ws, size_t ws_size,
                              hipStream_t stream) {
  (void)in_sizes; (void)n_in; (void)out_size; (void)ws_size;

  const float* feat  = (const float*)d_in[0];   // [8192,128] f32
  const int*   label = (const int*)d_in[1];     // [8192] int
  float*       out   = (float*)d_out;           // scalar

  // Workspace layout: [0, 2MB) f16 normalized features; then per-row partials.
  _Float16* f16  = (_Float16*)d_ws;
  float*    part = (float*)((char*)d_ws + (size_t)BATCH * DIM * sizeof(_Float16));

  circle_normalize<<<BATCH / 8, 256, 0, stream>>>(feat, f16);
  circle_main<<<(BATCH / 16) * COL_SPLIT / 4, 128, 0, stream>>>(f16, label, part);
  circle_finalize<<<1, 256, 0, stream>>>(part, out);
}
